// TopKTokenSelector_1786706395796
// MI455X (gfx1250) — matrix operand order repaired and checked by reading, not compiled, not run on hardware
//
#include <hip/hip_runtime.h>
#include <hip/hip_bf16.h>
#include <math.h>

#define B_  64
#define N_  1025
#define D_  768
#define H_  128
#define K_  256
#define NT  65          // ceil(1025/16) token tiles per batch
#define AST 776         // padded LDS stride in bf16 elems (388 dwords % 64 banks = 4)

typedef __attribute__((ext_vector_type(16))) __bf16          v16bf;
typedef __attribute__((ext_vector_type(8)))  float           v8f;
typedef __attribute__((ext_vector_type(8)))  unsigned short  u16x8;

union BF16x16 { v16bf v; u16x8 h[2]; };

__device__ __forceinline__ unsigned short f2bf(float f) {
  unsigned u = __float_as_uint(f);
  u += 0x7FFFu + ((u >> 16) & 1u);           // round-to-nearest-even
  return (unsigned short)(u >> 16);
}

// ---------- kernel 0: W1 (D x H, f32, row-major) -> w1t (H x D, bf16, K-contiguous) ----------
__global__ void __launch_bounds__(256) k_w1t(const float* __restrict__ w1,
                                             unsigned short* __restrict__ w1t) {
  int i = blockIdx.x * blockDim.x + threadIdx.x;   // i = n*D + k
  if (i >= D_ * H_) return;
  int n = i / D_;
  int k = i - n * D_;
  w1t[i] = f2bf(w1[k * H_ + n]);
}

// ---------- kernel 1: fused LN -> GEMM(WMMA bf16) -> GELU -> dot(w2) -> scores ----------
__global__ void __launch_bounds__(256) k_scores(
    const float* __restrict__ x,
    const float* __restrict__ lnw, const float* __restrict__ lnb,
    const unsigned short* __restrict__ w1t,
    const float* __restrict__ b1, const float* __restrict__ w2,
    const float* __restrict__ b2, float* __restrict__ scores) {

  __shared__ unsigned short Als[16 * AST];   // LN output tile, 16 tokens x 768 bf16 (padded)
  __shared__ float sp[8][16];                // per-wave partial scores

  const int b       = blockIdx.y;
  const int tokBase = blockIdx.x * 16;
  const int tid     = threadIdx.x;

  // ---- Phase 1: LayerNorm (16 threads per token, 48 cols each) ----
  {
    const int tok  = tid >> 4;
    const int sub  = tid & 15;
    const int gtok = tokBase + tok;
    const int col0 = sub * 48;
    if (gtok < N_) {
      float vals[48];
      const float4* xp = (const float4*)(x + ((size_t)b * N_ + gtok) * D_ + col0);
      float s = 0.f, s2 = 0.f;
      #pragma unroll
      for (int q = 0; q < 12; ++q) {
        float4 v4 = xp[q];
        vals[q*4+0] = v4.x; vals[q*4+1] = v4.y; vals[q*4+2] = v4.z; vals[q*4+3] = v4.w;
        s  += v4.x + v4.y + v4.z + v4.w;
        s2 += v4.x*v4.x + v4.y*v4.y + v4.z*v4.z + v4.w*v4.w;
      }
      // reduce across the 16 lanes of this token (masks 1,2,4,8 stay in the 16-lane half)
      #pragma unroll
      for (int m = 1; m < 16; m <<= 1) {
        s  += __shfl_xor(s,  m, 32);
        s2 += __shfl_xor(s2, m, 32);
      }
      float mu  = s  * (1.f / 768.f);
      float var = s2 * (1.f / 768.f) - mu * mu;
      float rs  = rsqrtf(var + 1e-5f);
      #pragma unroll
      for (int q = 0; q < 48; ++q) {
        int   c = col0 + q;
        float h = (vals[q] - mu) * rs * lnw[c] + lnb[c];
        Als[tok * AST + c] = f2bf(h);
      }
    } else {
      for (int q = 0; q < 48; ++q) Als[tok * AST + col0 + q] = 0;
    }
  }
  __syncthreads();

  // ---- Phase 2: WMMA GEMM, wave w handles N-tile w (H = 8 tiles of 16) ----
  const int wave = tid >> 5;
  const int lane = tid & 31;
  const int row  = lane & 15;
  const int half = lane >> 4;
  const int n    = wave * 16 + row;          // output column (fixed per lane)

  v8f acc = {};
  const unsigned short* bp = w1t + (size_t)n * D_ + half * 8;
  #pragma unroll 4
  for (int kb = 0; kb < 24; ++kb) {
    // A 16x32 bf16 layout (ISA 7.12.2): lane<16 holds K 0-7 & 16-23, lane>=16 holds K 8-15 & 24-31
    BF16x16 a, bm;
    a.h[0] = *(const u16x8*)(&Als[row * AST + kb * 32 + half * 8]);
    a.h[1] = *(const u16x8*)(&Als[row * AST + kb * 32 + 16 + half * 8]);
    // B 32x16 bf16: lane<16 holds K 0-7 & 16-23 of column n, lane>=16 holds K 8-15 & 24-31
    bm.h[0] = *(const u16x8*)(bp + kb * 32);
    bm.h[1] = *(const u16x8*)(bp + kb * 32 + 16);
    acc = __builtin_amdgcn_wmma_f32_16x16x32_bf16(false, a.v, false, bm.v,
                                                  (short)0, acc, false, false);
  }

  // ---- Phase 3: +b1, exact GELU, x w2[n], reduce across N ----
  {
    const float b1n = b1[n];
    const float w2n = w2[n];
    float part[8];
    #pragma unroll
    for (int r = 0; r < 8; ++r) {
      float v = acc[r] + b1n;
      float g = 0.5f * v * (1.0f + erff(v * 0.70710678118654752440f));
      part[r] = g * w2n;
    }
    #pragma unroll
    for (int m = 1; m < 16; m <<= 1) {
      #pragma unroll
      for (int r = 0; r < 8; ++r) part[r] += __shfl_xor(part[r], m, 32);
    }
    if (lane == 0) {
      for (int r = 0; r < 8; ++r) sp[wave][r] = part[r];
    }
    if (lane == 16) {
      for (int r = 0; r < 8; ++r) sp[wave][8 + r] = part[r];
    }
  }
  __syncthreads();

  if (tid < 16) {
    int gtok = tokBase + tid;
    if (gtok < N_) {
      float s = b2[0];
      #pragma unroll
      for (int w = 0; w < 8; ++w) s += sp[w][tid];
      scores[(size_t)b * N_ + gtok] = (gtok == 0) ? 1e9f : s;
    }
  }
}

// ---------- kernel 2: per-batch descending bitonic sort of 1024 (score, idx), take top 256 ----------
__global__ void __launch_bounds__(512) k_topk(const float* __restrict__ scores,
                                              float* __restrict__ outIdx,
                                              int* __restrict__ wsIdx) {
  __shared__ float ss[1024];
  __shared__ int   si[1024];
  const int b = blockIdx.x;
  const int t = threadIdx.x;

  for (int i = t; i < 1024; i += 512) {
    ss[i] = scores[(size_t)b * N_ + 1 + i];
    si[i] = i + 1;
  }
  for (int k = 2; k <= 1024; k <<= 1) {
    for (int j = k >> 1; j > 0; j >>= 1) {
      __syncthreads();
      int i   = ((t & ~(j - 1)) << 1) | (t & (j - 1));
      int ixj = i | j;
      float s0 = ss[i],  s1 = ss[ixj];
      int   i0 = si[i],  i1 = si[ixj];
      bool first = (s0 > s1) || (s0 == s1 && i0 < i1);  // s0 precedes in descending order
      bool up    = (i & k) == 0;
      if (up ? !first : first) {
        ss[i] = s1;  si[i] = i1;
        ss[ixj] = s0; si[ixj] = i0;
      }
    }
  }
  __syncthreads();
  if (t < K_) {
    outIdx[(size_t)b * 257 + 1 + t] = (float)si[t];
    wsIdx[b * 257 + 1 + t] = si[t];
  }
  if (t == 0) { outIdx[(size_t)b * 257] = 0.0f; wsIdx[b * 257] = 0; }
}

// ---------- kernel 3: gather selected rows of x ----------
__global__ void __launch_bounds__(192) k_gather(const float* __restrict__ x,
                                                const int* __restrict__ wsIdx,
                                                float* __restrict__ outX) {
  const int bid = blockIdx.x;
  const int b   = bid / 257;
  const int i   = bid - b * 257;
  const int src = wsIdx[b * 257 + i];
  const float4* spx = (const float4*)(x + ((size_t)b * N_ + src) * D_);
  float4*       dpx = (float4*)(outX + ((size_t)b * 257 + i) * D_);
  dpx[threadIdx.x] = spx[threadIdx.x];   // 192 * 16B = 768 floats
}

extern "C" void kernel_launch(void* const* d_in, const int* in_sizes, int n_in,
                              void* d_out, int out_size, void* d_ws, size_t ws_size,
                              hipStream_t stream) {
  const float* x   = (const float*)d_in[0];
  const float* lnw = (const float*)d_in[1];
  const float* lnb = (const float*)d_in[2];
  const float* w1  = (const float*)d_in[3];
  const float* b1  = (const float*)d_in[4];
  const float* w2  = (const float*)d_in[5];
  const float* b2  = (const float*)d_in[6];

  float* outX   = (float*)d_out;                         // [64,257,768]
  float* outIdx = outX + (size_t)B_ * 257 * D_;          // [64,257] (as float values)
  float* outSc  = outIdx + (size_t)B_ * 257;             // [64,1025]

  int* wsIdx = (int*)d_ws;                               // [64,257] int copy for gather
  size_t w1tOff = ((size_t)B_ * 257 * sizeof(int) + 255) & ~(size_t)255;
  unsigned short* w1t = (unsigned short*)((char*)d_ws + w1tOff);  // [128,768] bf16

  k_w1t   <<<dim3((D_ * H_ + 255) / 256), dim3(256), 0, stream>>>(w1, w1t);
  k_scores<<<dim3(NT, B_),                dim3(256), 0, stream>>>(x, lnw, lnb, w1t, b1, w2, b2, outSc);
  k_topk  <<<dim3(B_),                    dim3(512), 0, stream>>>(outSc, outIdx, wsIdx);
  k_gather<<<dim3(B_ * 257),              dim3(192), 0, stream>>>(x, wsIdx, outX);
}
